// Capsule_Net_80101140070969
// MI455X (gfx1250) — matrix operand order repaired and checked by reading, not compiled, not run on hardware
//
#include <hip/hip_runtime.h>
#include <hip/hip_bf16.h>
#include <math.h>

// ---------------------------------------------------------------------------
// CapsNet forward for MI455X (gfx1250, wave32, WMMA).
// Heavy lifting (primary-caps conv ~196 GFLOP, digit-caps GEMM, recon MLP)
// runs on v_wmma_f32_16x16x32_f16 with fp32 accumulation.
// Routing with r=1 and b_log=0 reduces to c=1/10 => single GEMM, no [B,1152,10,16].
// ---------------------------------------------------------------------------

typedef __attribute__((ext_vector_type(16))) _Float16 v16h;
typedef __attribute__((ext_vector_type(8)))  _Float16 v8h;
typedef __attribute__((ext_vector_type(8)))  float    v8f;

#define NB 512   // batch

// ---- WMMA fragment helpers (layouts per CDNA5 ISA 7.12.2) ------------------
// A 16x32 f16: lane holds row m=lane%16; half=lane/16:
//   VGPR0..3 : k = half*8 + 0..7     VGPR4..7 : k = 16 + half*8 + 0..7
__device__ __forceinline__ v16h load_fragA(const _Float16* __restrict__ p, int half) {
  union { v16h v; v8h h[2]; } u;
  u.h[0] = *(const v8h*)(p + half * 8);
  u.h[1] = *(const v8h*)(p + 16 + half * 8);
  return u.v;
}
// B 32x16 f16 (stored as Bt[N][K]): lane holds col n=lane%16; k = half*8 + 0..7, 16 + half*8 + 0..7
__device__ __forceinline__ v16h load_fragB(const _Float16* __restrict__ p, int half) {
  union { v16h v; v8h h[2]; } u;
  u.h[0] = *(const v8h*)(p + half * 8);
  u.h[1] = *(const v8h*)(p + 16 + half * 8);
  return u.v;
}
__device__ __forceinline__ v8f wmma16(v16h a, v16h b, v8f c) {
  return __builtin_amdgcn_wmma_f32_16x16x32_f16(false, a, false, b, (short)0, c,
                                                false, false);
}

// ---- prep kernels ----------------------------------------------------------
__global__ void k_cast_f16(const float* __restrict__ src, _Float16* __restrict__ dst, int n) {
  int t = blockIdx.x * blockDim.x + threadIdx.x;
  if (t < n) dst[t] = (_Float16)src[t];
}

// pc_w [O=256][C=256][9][9] fp32  ->  Wt f16 [kpos=81][O=256][C=256]
__global__ void k_prep_pcw(const float* __restrict__ src, _Float16* __restrict__ dst) {
  int t = blockIdx.x * blockDim.x + threadIdx.x;             // 81*256*256
  int c = t & 255, o = (t >> 8) & 255, kpos = t >> 16;
  dst[t] = (_Float16)src[(o * 256 + c) * 81 + kpos];
}

// W_dig [1152][10][16][8] fp32 -> W2t f16 [N=160][K=9216], N=(j,o), K=(i,d)
__global__ void k_prep_w2t(const float* __restrict__ src, _Float16* __restrict__ dst) {
  int t = blockIdx.x * blockDim.x + threadIdx.x;             // 160*9216
  int k = t % 9216, n = t / 9216;
  int i = k >> 3, d = k & 7, j = n >> 4, o = n & 15;
  dst[t] = (_Float16)src[((i * 10 + j) * 16 + o) * 8 + d];
}

// conv1_w [256][1][9][9] fp32 -> w1t fp32 [81][256]
__global__ void k_prep_w1t(const float* __restrict__ src, float* __restrict__ dst) {
  int t = blockIdx.x * blockDim.x + threadIdx.x;             // 81*256
  int ch = t & 255, kp = t >> 8;
  dst[t] = src[ch * 81 + kp];
}

// ---- conv1: direct fp32, writes NHWC f16 [512][20][20][256] ---------------
__global__ void k_conv1(const float* __restrict__ x, const float* __restrict__ w1t,
                        const float* __restrict__ bias, _Float16* __restrict__ H) {
  __shared__ float patch[81];
  int bid = blockIdx.x;                 // b*400 + y*20 + xo
  int b = bid / 400, rem = bid % 400;
  int y = rem / 20, xo = rem % 20;
  int ch = threadIdx.x;                 // 256
  if (ch < 81) {
    int ky = ch / 9, kx = ch % 9;
    patch[ch] = x[b * 784 + (y + ky) * 28 + (xo + kx)];
  }
  __syncthreads();
  float s = bias[ch];
#pragma unroll 9
  for (int k = 0; k < 81; ++k) s += patch[k] * w1t[k * 256 + ch];
  H[(size_t)bid * 256 + ch] = (_Float16)s;
}

// ---- primary-caps conv as implicit GEMM on WMMA ---------------------------
// M=18432 (b,y,x), N=256 (out ch), K=81*256.
// 32(M) x 64(N) tile per wave: 8 WMMAs per 12 b128 fragment loads per k-chunk.
__global__ void k_pcconv_wmma(const _Float16* __restrict__ H,   // [512][20][20][256]
                              const _Float16* __restrict__ Wt,  // [81][256][256]
                              const float* __restrict__ bias,   // [256]
                              float* __restrict__ O) {          // [18432][256]
  const int lane = threadIdx.x & 31;
  const int wave = blockIdx.x * (blockDim.x >> 5) + (threadIdx.x >> 5);
  const int tn = wave & 3;          // 4 n-tiles of 64
  const int tm = wave >> 2;         // 576 m-tiles of 32
  const int m0 = tm * 32, n0 = tn * 64;
  const int half = lane >> 4, lm = lane & 15;

  auto hrow = [&](int m) -> const _Float16* {
    int b = m / 36, pos = m % 36, y = pos / 6, xx = pos % 6;
    return H + (size_t)((b * 20 + 2 * y) * 20 + 2 * xx) * 256;
  };
  const _Float16* a0p = hrow(m0 + lm);
  const _Float16* a1p = hrow(m0 + 16 + lm);
  const _Float16* bp[4];
#pragma unroll
  for (int nn = 0; nn < 4; ++nn) bp[nn] = Wt + (size_t)(n0 + nn * 16 + lm) * 256;

  v8f acc[2][4] = {};
  for (int ky = 0; ky < 9; ++ky)
    for (int kx = 0; kx < 9; ++kx) {
      const int kpos = ky * 9 + kx;
      const _Float16* ar0 = a0p + (ky * 20 + kx) * 256;
      const _Float16* ar1 = a1p + (ky * 20 + kx) * 256;
      // prefetch next kernel position's activation rows (global_prefetch_b8)
      if (kpos < 80) {
        int nky = (kpos + 1) / 9, nkx = (kpos + 1) % 9;
        __builtin_prefetch(a0p + (nky * 20 + nkx) * 256, 0, 3);
        __builtin_prefetch(a1p + (nky * 20 + nkx) * 256, 0, 3);
      }
#pragma unroll
      for (int cc = 0; cc < 8; ++cc) {      // 256 channels / 32
        v16h A0 = load_fragA(ar0 + cc * 32, half);
        v16h A1 = load_fragA(ar1 + cc * 32, half);
#pragma unroll
        for (int nn = 0; nn < 4; ++nn) {
          v16h Bn = load_fragB(bp[nn] + (size_t)kpos * 65536 + cc * 32, half);
          acc[0][nn] = wmma16(A0, Bn, acc[0][nn]);
          acc[1][nn] = wmma16(A1, Bn, acc[1][nn]);
        }
      }
    }
  float bn[4];
#pragma unroll
  for (int nn = 0; nn < 4; ++nn) bn[nn] = bias[n0 + nn * 16 + lm];
#pragma unroll
  for (int j = 0; j < 8; ++j) {
    int mA = m0 + half * 8 + j, mB = mA + 16;
#pragma unroll
    for (int nn = 0; nn < 4; ++nn) {
      O[(size_t)mA * 256 + n0 + nn * 16 + lm] = acc[0][nn][j] + bn[nn];
      O[(size_t)mB * 256 + n0 + nn * 16 + lm] = acc[1][nn][j] + bn[nn];
    }
  }
}

// ---- squash primary caps -> u f16 [512][9216], k = (cap*36+pos)*8 + d -----
__global__ void k_squash_primary(const float* __restrict__ P,   // [512*36][256]
                                 _Float16* __restrict__ U) {
  int t = blockIdx.x * blockDim.x + threadIdx.x;                // 512*36*32
  int cap = t & 31, pos = (t >> 5) % 36, b = t / (36 * 32);
  const float* v = P + (size_t)(b * 36 + pos) * 256 + cap * 8;
  float xv[8], n2 = 0.f;
#pragma unroll
  for (int d = 0; d < 8; ++d) { xv[d] = v[d]; n2 += xv[d] * xv[d]; }
  float scale = sqrtf(n2) / (1.f + n2);     // == (1/n)*n^2/(1+n^2)
  _Float16* u = U + ((size_t)b * 1152 + cap * 36 + pos) * 8;
#pragma unroll
  for (int d = 0; d < 8; ++d) u[d] = (_Float16)(xv[d] * scale);
}

// ---- generic WMMA GEMM: C[M][N] = A[M][K] * Bt[N][K]^T (+bias, act) -------
// 32(M) x 16(N) tile per wave. EPI: 0=none 1=relu 2=sigmoid.
template <int EPI, bool HALF_OUT>
__global__ void k_gemm_wmma(const _Float16* __restrict__ A,
                            const _Float16* __restrict__ Bt,
                            const float* __restrict__ bias,
                            void* __restrict__ C, int M, int N, int K) {
  const int lane = threadIdx.x & 31;
  const int wave = blockIdx.x * (blockDim.x >> 5) + (threadIdx.x >> 5);
  const int ntile = N >> 4;
  if (wave >= (M >> 5) * ntile) return;     // uniform per wave: EXEC stays full
  const int tn = wave % ntile, tm = wave / ntile;
  const int m0 = tm * 32, n0 = tn * 16;
  const int half = lane >> 4, lm = lane & 15;

  const _Float16* ar0 = A + (size_t)(m0 + lm) * K;
  const _Float16* ar1 = A + (size_t)(m0 + 16 + lm) * K;
  const _Float16* br  = Bt + (size_t)(n0 + lm) * K;

  v8f acc0 = {}, acc1 = {};
  for (int k = 0; k < K; k += 32) {
    v16h a0 = load_fragA(ar0 + k, half);
    v16h a1 = load_fragA(ar1 + k, half);
    v16h bb = load_fragB(br + k, half);
    acc0 = wmma16(a0, bb, acc0);
    acc1 = wmma16(a1, bb, acc1);
  }
  const int n = n0 + lm;
  float bn = bias ? bias[n] : 0.f;
#pragma unroll
  for (int j = 0; j < 8; ++j) {
    float v0 = acc0[j] + bn, v1 = acc1[j] + bn;
    if (EPI == 1) { v0 = v0 > 0.f ? v0 : 0.f; v1 = v1 > 0.f ? v1 : 0.f; }
    if (EPI == 2) { v0 = 1.f / (1.f + expf(-v0)); v1 = 1.f / (1.f + expf(-v1)); }
    size_t i0 = (size_t)(m0 + half * 8 + j) * N + n;
    size_t i1 = (size_t)(m0 + 16 + half * 8 + j) * N + n;
    if (HALF_OUT) { ((_Float16*)C)[i0] = (_Float16)v0; ((_Float16*)C)[i1] = (_Float16)v1; }
    else          { ((float*)C)[i0] = v0;              ((float*)C)[i1] = v1; }
  }
}

// ---- routing (c=1/10), squash digit caps, argmax mask ---------------------
__global__ void k_route_mask(const float* __restrict__ SP,  // [512][160] pre-scale
                             float* __restrict__ Sout,      // d_out s [512][160]
                             _Float16* __restrict__ Rin) {  // masked [512][160] f16
  __shared__ float sv[160];
  __shared__ float n2s[10];
  __shared__ int jbest;
  int b = blockIdx.x, t = threadIdx.x;     // 160 threads
  int j = t >> 4;
  float v = 0.1f * SP[b * 160 + t];        // c = softmax(0) = 1/10
  sv[t] = v * v;
  __syncthreads();
  if (t < 10) {
    float s = 0.f;
    for (int k = 0; k < 16; ++k) s += sv[t * 16 + k];
    n2s[t] = s;
  }
  __syncthreads();
  if (t == 0) {
    int bj = 0; float bv = n2s[0];
    for (int k = 1; k < 10; ++k) if (n2s[k] > bv) { bv = n2s[k]; bj = k; }
    jbest = bj;                            // |squash(s)| monotonic in |s|
  }
  __syncthreads();
  float n2 = n2s[j];
  float so = v * (sqrtf(n2) / (1.f + n2));
  Sout[b * 160 + t] = so;
  Rin[b * 160 + t] = (_Float16)((j == jbest) ? so : 0.f);
}

// ---------------------------------------------------------------------------
extern "C" void kernel_launch(void* const* d_in, const int* in_sizes, int n_in,
                              void* d_out, int out_size, void* d_ws, size_t ws_size,
                              hipStream_t stream) {
  const float* x    = (const float*)d_in[0];
  const float* c1w  = (const float*)d_in[1];
  const float* c1b  = (const float*)d_in[2];
  const float* pcw  = (const float*)d_in[3];
  const float* pcb  = (const float*)d_in[4];
  const float* wdig = (const float*)d_in[5];
  const float* d1w  = (const float*)d_in[6];
  const float* d1b  = (const float*)d_in[7];
  const float* d2w  = (const float*)d_in[8];
  const float* d2b  = (const float*)d_in[9];
  const float* d3w  = (const float*)d_in[10];
  const float* d3b  = (const float*)d_in[11];
  float* out_s = (float*)d_out;            // [512][160]
  float* out_r = out_s + NB * 160;         // [512][784]

  // workspace partition (256B aligned), total ~152 MB
  char* ws = (char*)d_ws;
  size_t off = 0;
  auto take = [&](size_t bytes) { char* p = ws + off; off += (bytes + 255) & ~(size_t)255; return p; };
  _Float16* H16  = (_Float16*)take((size_t)NB * 400 * 256 * 2);  // conv1 out NHWC f16
  _Float16* WPC  = (_Float16*)take((size_t)81 * 256 * 256 * 2);  // pc weights [kpos][o][c]
  float*    PCO  = (float*)   take((size_t)NB * 36 * 256 * 4);   // pc-conv out NHWC f32
  _Float16* U16  = (_Float16*)take((size_t)NB * 9216 * 2);       // squashed primary caps
  _Float16* W2T  = (_Float16*)take((size_t)160 * 9216 * 2);      // digit weights [N][K]
  float*    SPRE = (float*)   take((size_t)NB * 160 * 4);        // digit GEMM out
  _Float16* RIN  = (_Float16*)take((size_t)NB * 160 * 2);        // masked recon input
  _Float16* D1W  = (_Float16*)take((size_t)512 * 160 * 2);
  _Float16* D2W  = (_Float16*)take((size_t)1024 * 512 * 2);
  _Float16* D3W  = (_Float16*)take((size_t)784 * 1024 * 2);
  _Float16* H1   = (_Float16*)take((size_t)NB * 512 * 2);
  _Float16* H2   = (_Float16*)take((size_t)NB * 1024 * 2);
  float*    W1T  = (float*)   take((size_t)81 * 256 * 4);
  (void)ws_size; (void)in_sizes; (void)n_in; (void)out_size;

  // weight prep
  k_cast_f16<<<(512 * 160 + 255) / 256, 256, 0, stream>>>(d1w, D1W, 512 * 160);
  k_cast_f16<<<(1024 * 512 + 255) / 256, 256, 0, stream>>>(d2w, D2W, 1024 * 512);
  k_cast_f16<<<(784 * 1024 + 255) / 256, 256, 0, stream>>>(d3w, D3W, 784 * 1024);
  k_prep_pcw<<<(81 * 256 * 256) / 256, 256, 0, stream>>>(pcw, WPC);
  k_prep_w2t<<<(160 * 9216) / 256, 256, 0, stream>>>(wdig, W2T);
  k_prep_w1t<<<(81 * 256) / 256, 256, 0, stream>>>(c1w, W1T);

  // conv1 (direct fp32 -> NHWC f16)
  k_conv1<<<NB * 400, 256, 0, stream>>>(x, W1T, c1b, H16);

  // primary-caps conv: 576 m-tiles * 4 n-tiles(64) = 2304 waves, 8 waves/block
  k_pcconv_wmma<<<288, 256, 0, stream>>>(H16, WPC, pcb, PCO);

  // squash -> u f16
  k_squash_primary<<<(NB * 36 * 32) / 256, 256, 0, stream>>>(PCO, U16);

  // digit caps: [512 x 9216] x [9216 x 160]
  k_gemm_wmma<0, false><<<(16 * 10) / 8, 256, 0, stream>>>(U16, W2T, nullptr, SPRE,
                                                           NB, 160, 9216);
  // routing (c=1/10) + squash + argmax mask
  k_route_mask<<<NB, 160, 0, stream>>>(SPRE, out_s, RIN);

  // recon MLP
  k_gemm_wmma<1, true><<<(16 * 32) / 8, 256, 0, stream>>>(RIN, D1W, d1b, H1, NB, 512, 160);
  k_gemm_wmma<1, true><<<(16 * 64) / 8, 256, 0, stream>>>(H1, D2W, d2b, H2, NB, 1024, 512);
  k_gemm_wmma<2, false><<<(16 * 49) / 8, 256, 0, stream>>>(H2, D3W, d3b, out_r, NB, 784, 1024);
}
